// DefaultSegmentLinear_68796786147366
// MI455X (gfx1250) — 8-bit matrix operand order repaired and checked by reading, compiled, not run on hardware
//
#include <hip/hip_runtime.h>


typedef __attribute__((ext_vector_type(16))) int   v16i;
typedef __attribute__((ext_vector_type(8)))  float v8f;

#if defined(__has_builtin)
#if __has_builtin(__builtin_amdgcn_cvt_pk_fp8_f32)
#define HAVE_HW_FP8_CVT 1
#endif
#endif

__device__ __forceinline__ unsigned int f32_to_e4m3_byte(float x) {
    unsigned int u    = __float_as_uint(x);
    unsigned int sign = (u >> 24) & 0x80u;
    unsigned int au   = u & 0x7FFFFFFFu;
    float ax = __uint_as_float(au);
    if (ax > 448.0f) { ax = 448.0f; au = __float_as_uint(ax); }
    int e = (int)(au >> 23) - 127;
    if (e < -6) {
        int q = (int)rintf(ax * 512.0f);
        if (q >= 8) return sign | 0x08u;
        return sign | (unsigned int)q;
    }
    unsigned int mant = au & 0x7FFFFFu;
    unsigned int lsb  = (mant >> 20) & 1u;
    mant += 0x7FFFFu + lsb;
    if (mant & 0x800000u) { mant = 0u; e += 1; }
    if (e > 8) return sign | 0x7Eu;
    unsigned int m3 = (mant >> 20) & 0x7u;
    return sign | ((unsigned int)(e + 7) << 3) | m3;
}

__device__ __forceinline__ float clamp448(float v) {
    return fminf(fmaxf(v, -448.0f), 448.0f);
}

__device__ __forceinline__ unsigned int quant4_e4m3(float4 v, float inv) {
#ifdef HAVE_HW_FP8_CVT
    float a = clamp448(v.x * inv), b = clamp448(v.y * inv);
    float c = clamp448(v.z * inv), d = clamp448(v.w * inv);
    int p = 0;
    p = __builtin_amdgcn_cvt_pk_fp8_f32(a, b, p, false);
    p = __builtin_amdgcn_cvt_pk_fp8_f32(c, d, p, true);
    return (unsigned int)p;
#else
    return  f32_to_e4m3_byte(v.x * inv)
         | (f32_to_e4m3_byte(v.y * inv) << 8)
         | (f32_to_e4m3_byte(v.z * inv) << 16)
         | (f32_to_e4m3_byte(v.w * inv) << 24);
#endif
}

__global__ __launch_bounds__(256) void quant_e4m3_kernel(
    const float* __restrict__ src, uint2* __restrict__ dst,
    long long n8, const float* __restrict__ scales, long long elems_per_chunk)
{
    long long i8 = (long long)blockIdx.x * blockDim.x + threadIdx.x;
    if (i8 >= n8) return;
    long long i = i8 * 8;
    int chunk = (int)(i / elems_per_chunk);
    float inv = 1.0f / scales[chunk];
    const float4* s4 = reinterpret_cast<const float4*>(src) + i8 * 2;
    float4 v0 = s4[0];
    float4 v1 = s4[1];
    uint2 p;
    p.x = quant4_e4m3(v0, inv);
    p.y = quant4_e4m3(v1, inv);
    dst[i8] = p;
}

__device__ __forceinline__ void async_copy_b128(unsigned lds_addr,
                                                const unsigned char* gptr) {
    asm volatile("global_load_async_to_lds_b128 %0, %1, off"
                 :: "v"(lds_addr), "v"((unsigned long long)(size_t)gptr)
                 : "memory");
}
__device__ __forceinline__ void wait_asynccnt0() {
    asm volatile("s_wait_asynccnt 0x0" ::: "memory");
}

__device__ __forceinline__ v16i load_a_frag(const unsigned char* p) {
    v16i a;
    const int2* ap = reinterpret_cast<const int2*>(p);
    #pragma unroll
    for (int j = 0; j < 8; ++j) {
        int2 t = ap[4 * (j >> 1) + (j & 1)];
        a[2 * j] = t.x; a[2 * j + 1] = t.y;
    }
    return a;
}
__device__ __forceinline__ v16i load_b_frag(const unsigned char* p) {
    v16i b;
    const int4* bp = reinterpret_cast<const int4*>(p);
    #pragma unroll
    for (int j = 0; j < 4; ++j) {
        int4 q = bp[2 * j];
        b[4 * j] = q.x; b[4 * j + 1] = q.y;
        b[4 * j + 2] = q.z; b[4 * j + 3] = q.w;
    }
    return b;
}

#define KSTEP 128
#define NBLK  128
#define STAGE_BYTES (NBLK * KSTEP)

__global__ __launch_bounds__(256) void fp8_gemm_kernel(
    const unsigned char* __restrict__ xq,
    const unsigned char* __restrict__ wq,
    const float* __restrict__ bias,
    const float* __restrict__ in_scale,
    const float* __restrict__ w_scales,
    float* __restrict__ out,
    int M, int N, int K)
{
    __shared__ unsigned char smem[2][STAGE_BYTES];

    const int tid    = threadIdx.x;
    const int lane   = tid & 31;
    const int wave   = tid >> 5;
    const int lane16 = lane & 15;
    const int laneHi = lane >> 4;

    const int nBlock = blockIdx.x * NBLK;
    const int nWave  = (wave & 1) * 64;
    const int mWave  = blockIdx.y * 128 + (wave >> 1) * 32;

    const unsigned char* aRow0 = xq + (size_t)(mWave + lane16) * K + (laneHi ? 16 : 0);
    const unsigned char* aRow1 = aRow0 + (size_t)16 * K;

    const int cpRow  = tid >> 1;
    const int cpHalf = (tid & 1) * 64;
    const unsigned char* bCopySrc = wq + (size_t)(nBlock + cpRow) * K + cpHalf;
    const unsigned ldsBase  = (unsigned)(size_t)(void*)&smem[0][0];
    const unsigned cpLdsOff = (unsigned)(cpRow * KSTEP + cpHalf);

    const unsigned bFragOff = (unsigned)((nWave + lane16) * KSTEP + laneHi * 16);

    v8f acc[2][4] = {};

    const int KT = K / KSTEP;

    #pragma unroll
    for (int j = 0; j < 4; ++j)
        async_copy_b128(ldsBase + cpLdsOff + 16 * j, bCopySrc + 16 * j);

    v16i aP0 = load_a_frag(aRow0);
    v16i aP1 = load_a_frag(aRow1);
    v16i aQ0 = {}, aQ1 = {};

    auto k_iter = [&](int kt, v16i& aU0, v16i& aU1,
                              v16i& aN0, v16i& aN1) __attribute__((always_inline)) {
        const int k   = kt * KSTEP;
        const int cur = kt & 1;

        wait_asynccnt0();
        __syncthreads();

        if (kt + 1 < KT) {
            const unsigned char* src = bCopySrc + (size_t)(k + KSTEP);
            const unsigned dstOff = ldsBase + (1 - cur) * STAGE_BYTES + cpLdsOff;
            #pragma unroll
            for (int j = 0; j < 4; ++j)
                async_copy_b128(dstOff + 16 * j, src + 16 * j);
            aN0 = load_a_frag(aRow0 + k + KSTEP);
            aN1 = load_a_frag(aRow1 + k + KSTEP);
        }

        const unsigned char* bStage = &smem[cur][0];
        v16i b0 = load_b_frag(bStage + bFragOff);
        v16i b1;
        #pragma unroll
        for (int t = 0; t < 4; ++t) {
            if (t + 1 < 4) {
                v16i nb = load_b_frag(bStage + bFragOff +
                                      (unsigned)((t + 1) * 16 * KSTEP));
                if (t & 1) b0 = nb; else b1 = nb;
            }
            const v16i bt = (t & 1) ? b1 : b0;
            acc[0][t] = __builtin_amdgcn_wmma_f32_16x16x128_fp8_fp8(
                aU0, bt, (short)0, acc[0][t], false, false);
            acc[1][t] = __builtin_amdgcn_wmma_f32_16x16x128_fp8_fp8(
                aU1, bt, (short)0, acc[1][t], false, false);
        }
    };

    for (int kt = 0; kt < KT; kt += 2) {
        k_iter(kt,     aP0, aP1, aQ0, aQ1);
        k_iter(kt + 1, aQ0, aQ1, aP0, aP1);
    }

    const int chunkSize = N >> 2;
    const float sIn = in_scale[0];

    #pragma unroll
    for (int t = 0; t < 4; ++t) {
        const int n   = nBlock + nWave + t * 16 + lane16;
        const float s  = sIn * w_scales[n / chunkSize];
        const float bv = bias[n];
        #pragma unroll
        for (int st = 0; st < 2; ++st) {
            #pragma unroll
            for (int r = 0; r < 8; ++r) {
                const int m = mWave + st * 16 + r + laneHi * 8;
                out[(size_t)m * N + n] = acc[st][t][r] * s + bv;
            }
        }
    }
}

extern "C" void kernel_launch(void* const* d_in, const int* in_sizes, int n_in,
                              void* d_out, int out_size, void* d_ws, size_t ws_size,
                              hipStream_t stream) {
    const float* x        = (const float*)d_in[0];
    const float* w        = (const float*)d_in[1];
    const float* bias     = (const float*)d_in[2];
    const float* in_scale = (const float*)d_in[3];
    const float* w_scales = (const float*)d_in[4];
    float* out = (float*)d_out;

    const long long xElems = in_sizes[0];
    const int OUT = in_sizes[2];
    const int IN  = (int)((long long)in_sizes[1] / OUT);
    const int M   = (int)(xElems / IN);

    unsigned char* xq = (unsigned char*)d_ws;
    unsigned char* wq = xq + (size_t)M * IN;

    {
        long long n8 = xElems / 8;
        int blocks = (int)((n8 + 255) / 256);
        quant_e4m3_kernel<<<blocks, 256, 0, stream>>>(
            x, (uint2*)xq, n8, in_scale, xElems);
    }
    {
        long long wElems = (long long)OUT * IN;
        long long n8 = wElems / 8;
        long long elemsPerChunk = (long long)(OUT / 4) * IN;
        int blocks = (int)((n8 + 255) / 256);
        quant_e4m3_kernel<<<blocks, 256, 0, stream>>>(
            w, (uint2*)wq, n8, w_scales, elemsPerChunk);
    }
    {
        dim3 grid(OUT / NBLK, M / 128);
        fp8_gemm_kernel<<<grid, 256, 0, stream>>>(
            xq, wq, bias, in_scale, w_scales, out, M, OUT, IN);
    }
}
